// Attention_40999757807902
// MI455X (gfx1250) — compile-verified, matrix operand order checked by reading, not run on hardware
//
#include <hip/hip_runtime.h>
#include <hip/hip_bf16.h>

// ---------------------------------------------------------------------------
// Types for CDNA5 WMMA (wave32, v_wmma_f32_16x16x32_bf16) + TDM descriptors
// ---------------------------------------------------------------------------
typedef __bf16 bf16;
typedef __attribute__((ext_vector_type(16))) __bf16 v16bf;
typedef __attribute__((ext_vector_type(8)))  float  v8f;
typedef __attribute__((ext_vector_type(4)))  unsigned int u32x4;
typedef __attribute__((ext_vector_type(8)))  int i32x8;
typedef __attribute__((ext_vector_type(4)))  int i32x4;

#define B_SZ   2
#define S_LEN  2048
#define DMODEL 4096
#define NHEAD  32
#define HDIM   128
#define NELEM  (4096 * 4096)          // elements per (B*S,D) tensor / weight

#define LDA 72                        // padded LDS row (bf16): 144B stride, bank-conflict free

__device__ __forceinline__ v8f wmma_bf16(v16bf a, v16bf b, v8f c) {
  // D = A(16x32 bf16) * B(32x16 bf16) + C(16x16 f32)
  return __builtin_amdgcn_wmma_f32_16x16x32_bf16(
      /*neg_a=*/false, a, /*neg_b=*/false, b,
      /*c_mod=*/(short)0, c, /*reuse_a=*/false, /*reuse_b=*/false);
}

// Load a 16x32 bf16 fragment in the ISA A/B layout:
//   lane<16 : row = row0+lane,    K = k0+{0..7, 16..23}
//   lane>=16: row = row0+lane-16, K = k0+{8..15, 24..31}
__device__ __forceinline__ v16bf load_frag(const bf16* base, int ld, int row0, int k0) {
  int lane = threadIdx.x & 31;
  const bf16* p = base + (size_t)(row0 + (lane & 15)) * ld + k0 + ((lane >> 4) << 3);
  union { v16bf v; uint4 q[2]; } f;
  f.q[0] = *(const uint4*)(p);
  f.q[1] = *(const uint4*)(p + 16);
  return f.v;
}

// 16-lane-half reductions (rows 0-7 live in lanes 0-15, rows 8-15 in 16-31)
__device__ __forceinline__ float half_red_max(float v) {
  v = fmaxf(v, __shfl_xor(v, 1, 32));
  v = fmaxf(v, __shfl_xor(v, 2, 32));
  v = fmaxf(v, __shfl_xor(v, 4, 32));
  v = fmaxf(v, __shfl_xor(v, 8, 32));
  return v;
}
__device__ __forceinline__ float half_red_sum(float v) {
  v += __shfl_xor(v, 1, 32);
  v += __shfl_xor(v, 2, 32);
  v += __shfl_xor(v, 4, 32);
  v += __shfl_xor(v, 8, 32);
  return v;
}

// ---------------------------------------------------------------------------
// TDM: DMA a 32x128 bf16 tile (row stride DMODEL) from global into LDS.
// D# per CDNA5 ISA ch.8: group0 = {count, lds_addr, global_addr, type=2},
// group1 = {data_size=2B, tensor_dim0/1, tile_dim0/1, tensor_dim0_stride}.
// Issued by one wave; completion via TENSORcnt (in-order per wave).
// ---------------------------------------------------------------------------
__device__ __forceinline__ void tdm_load_k_tile(const bf16* gsrc, bf16* ldsDst) {
  unsigned lds_off = (unsigned)(uintptr_t)ldsDst;   // flat LDS addr[31:0] == LDS offset
  unsigned long long ga = (unsigned long long)(uintptr_t)gsrc;
  u32x4 g0;
  g0[0] = 1u;                                                  // count=1 (valid user D#)
  g0[1] = lds_off;                                             // lds_addr (bytes)
  g0[2] = (unsigned)(ga & 0xFFFFFFFFu);                        // global_addr[31:0]
  g0[3] = (unsigned)((ga >> 32) & 0x01FFFFFFu) | (2u << 30);   // global_addr[56:32] | type=2
  i32x8 g1;
  g1[0] = 0x00010000;              // data_size=1 (2 bytes); no multicast/pad/iterate
  g1[1] = (int)(128u << 16);       // tensor_dim0 = 128      (bits 79:48)
  g1[2] = (int)(2048u << 16);      // tensor_dim1 = 2048     (bits 111:80)
  g1[3] = (int)(128u << 16);       // tile_dim0  = 128       (bits 127:112)
  g1[4] = 32;                      // tile_dim1  = 32        (bits 143:128)
  g1[5] = DMODEL;                  // tensor_dim0_stride     (bits 207:160)
  g1[6] = 0;
  g1[7] = 0;
  i32x4 z4 = {0, 0, 0, 0};         // groups 2/3 unused (2D tile)
  i32x8 z8 = {0, 0, 0, 0, 0, 0, 0, 0};
  __builtin_amdgcn_tensor_load_to_lds(g0, g1, z4, z4, z8, 0);
}

// ---------------------------------------------------------------------------
// fp32 -> bf16 downconvert, 8 elems/thread (b128 in/out)
// ---------------------------------------------------------------------------
__global__ __launch_bounds__(256) void cvt_f32_bf16(const float* __restrict__ in,
                                                    bf16* __restrict__ out, int n8) {
  int i = blockIdx.x * 256 + threadIdx.x;
  if (i < n8) {
    const float4* p = (const float4*)in + (size_t)i * 2;
    float4 a = p[0], b = p[1];
    union { bf16 h[8]; uint4 q; } u;
    u.h[0] = (bf16)a.x; u.h[1] = (bf16)a.y; u.h[2] = (bf16)a.z; u.h[3] = (bf16)a.w;
    u.h[4] = (bf16)b.x; u.h[5] = (bf16)b.y; u.h[6] = (bf16)b.z; u.h[7] = (bf16)b.w;
    *((uint4*)out + i) = u.q;
  }
}

// ---------------------------------------------------------------------------
// C(MxN) = A(MxK) * W(NxK)^T   -- bf16 in, f32 accumulate.
// Block: 256 thr = 8 waves. Block tile 128x128, K-step 64, LDS double-buffered.
// Wave grid 2(M) x 4(N); wave tile 64x32 = 4x2 fragments (8 f32 accumulators).
// ---------------------------------------------------------------------------
template <typename OUT_T>
__global__ __launch_bounds__(256) void gemm_xwT(const bf16* __restrict__ A,
                                                const bf16* __restrict__ W,
                                                OUT_T* __restrict__ C,
                                                int M, int N, int K) {
  __shared__ __align__(16) bf16 lA[2][128][LDA];
  __shared__ __align__(16) bf16 lB[2][128][LDA];

  const int tid  = threadIdx.x;
  const int lane = tid & 31;
  const int wid  = tid >> 5;
  const int mw   = wid >> 2;          // 0..1
  const int nw   = wid & 3;           // 0..3
  const int m0w  = mw * 64;
  const int n0w  = nw * 32;
  const int aRow0 = blockIdx.y * 128;
  const int bRow0 = blockIdx.x * 128;

  v8f acc[4][2] = {};
  uint4 ra[4], rb[4];

  // global -> regs for K-chunk k0 (128 rows x 64 cols, 8 bf16 per chunk)
  auto gload = [&](int k0) {
#pragma unroll
    for (int i = 0; i < 4; ++i) {
      int idx = tid + i * 256;                  // 1024 chunks
      int row = idx >> 3, c = (idx & 7) << 3;
      ra[i] = *(const uint4*)(A + (size_t)(aRow0 + row) * K + k0 + c);
      rb[i] = *(const uint4*)(W + (size_t)(bRow0 + row) * K + k0 + c);
    }
    // pull the chunk after next into L2 (global_prefetch_b8, no LOADcnt cost)
    if (k0 + 128 < K) {
      int row = tid >> 3, c = (tid & 7) << 3;
      __builtin_prefetch(A + (size_t)(aRow0 + row) * K + k0 + 128 + c, 0, 1);
      __builtin_prefetch(W + (size_t)(bRow0 + row) * K + k0 + 128 + c, 0, 1);
    }
  };
  // regs -> LDS buffer
  auto lstore = [&](int buf) {
#pragma unroll
    for (int i = 0; i < 4; ++i) {
      int idx = tid + i * 256;
      int row = idx >> 3, c = (idx & 7) << 3;
      *(uint4*)&lA[buf][row][c] = ra[i];
      *(uint4*)&lB[buf][row][c] = rb[i];
    }
  };

  const int KT = K >> 6;                        // K-steps of 64
  gload(0);
  lstore(0);
  __syncthreads();

  for (int kt = 0; kt < KT; ++kt) {
    const int cur = kt & 1;
    if (kt + 1 < KT) gload((kt + 1) << 6);      // prefetch next K-chunk

#pragma unroll
    for (int kc = 0; kc < 64; kc += 32) {
      v16bf af[4], bf[2];
#pragma unroll
      for (int m = 0; m < 4; ++m) af[m] = load_frag(&lA[cur][0][0], LDA, m0w + m * 16, kc);
#pragma unroll
      for (int n = 0; n < 2; ++n) bf[n] = load_frag(&lB[cur][0][0], LDA, n0w + n * 16, kc);
#pragma unroll
      for (int m = 0; m < 4; ++m)
#pragma unroll
        for (int n = 0; n < 2; ++n)
          acc[m][n] = wmma_bf16(af[m], bf[n], acc[m][n]);
    }

    __syncthreads();                            // all reads of lds done
    if (kt + 1 < KT) {
      lstore(cur ^ 1);
      __syncthreads();                          // buffer ready for next iter
    }
  }

  // C/D layout: VGPR r, lane<16 -> (M=r, N=lane); lane>=16 -> (M=r+8, N=lane-16)
#pragma unroll
  for (int m = 0; m < 4; ++m)
#pragma unroll
    for (int n = 0; n < 2; ++n)
#pragma unroll
      for (int r = 0; r < 8; ++r) {
        int row = aRow0 + m0w + m * 16 + ((lane >> 4) << 3) + r;
        int col = bRow0 + n0w + n * 16 + (lane & 15);
        C[(size_t)row * N + col] = (OUT_T)acc[m][n][r];
      }
}

// ---------------------------------------------------------------------------
// RoPE over (B,S,H,HD) bf16, cos/sin are (S, HD/2) fp32
// ---------------------------------------------------------------------------
__global__ __launch_bounds__(256) void rope_bf16(bf16* __restrict__ t,
                                                 const float* __restrict__ cv,
                                                 const float* __restrict__ sv) {
  int idx = blockIdx.x * 256 + threadIdx.x;   // B*S*H*64 threads
  int i = idx & 63;
  int h = (idx >> 6) & 31;
  int s = (idx >> 11) & 2047;
  int b = idx >> 22;
  size_t off = ((size_t)(b * S_LEN + s)) * DMODEL + h * HDIM + 2 * i;
  float tr = (float)t[off], ti = (float)t[off + 1];
  float c = cv[s * 64 + i], sn = sv[s * 64 + i];
  t[off]     = (bf16)(tr * c - ti * sn);
  t[off + 1] = (bf16)(tr * sn + ti * c);
}

// ---------------------------------------------------------------------------
// Flash attention (causal). Block = 128 thr (4 waves); each wave one 16-row
// q-tile; block covers 64 q rows of one (b,h). K tiles stream through a
// double-buffered LDS region via the Tensor Data Mover: wave 0 issues the DMA
// for tile kt+1, then s_wait_tensorcnt(1) guarantees tile kt has landed
// (tensor ops complete in order per wave) while kt+1 streams behind the WMMAs.
// V staged transposed so PV B-fragments are contiguous ds_read_b128.
// ---------------------------------------------------------------------------
__global__ __launch_bounds__(128) void flash_attn(const bf16* __restrict__ Q,
                                                  const bf16* __restrict__ K,
                                                  const bf16* __restrict__ V,
                                                  bf16* __restrict__ O) {
  __shared__ __align__(16) bf16 ldsK[2][32 * 128]; // [buf][kk][d] (TDM destination)
  __shared__ __align__(16) bf16 ldsVt[128 * 32];   // [d][kk]
  __shared__ __align__(16) bf16 ldsP[4 * 16 * 32]; // per-wave P scratch

  int tid  = threadIdx.x;
  int lane = tid & 31;
  int wid  = tid >> 5;

  // scalar (SGPR) wave-0 test: tensor ops ignore EXEC, so the guard must be
  // a scalar branch, not lane predication.
  const bool leaderWave = (__builtin_amdgcn_readfirstlane(tid) >> 5) == 0;

  int blk  = blockIdx.x;                 // B * NHEAD * (S/64)
  int qblk = blk & 31;                   // S/64 = 32
  int h    = (blk >> 5) & 31;
  int b    = blk >> 10;
  int qb   = qblk * 64;
  int qr0  = qb + wid * 16;

  const size_t bh = ((size_t)b * S_LEN) * DMODEL + (size_t)h * HDIM;

  // Preload the wave's q A-fragments (16 x 128 -> 4 frags along d)
  v16bf qa[4];
#pragma unroll
  for (int dk = 0; dk < 4; ++dk) qa[dk] = load_frag(Q + bh, DMODEL, qr0, dk * 32);

  v8f   oacc[8] = {};
  float mrow[8], lrow[8];
#pragma unroll
  for (int r = 0; r < 8; ++r) { mrow[r] = -3.0e38f; lrow[r] = 0.0f; }

  const float scale = 0.08838834764831845f;   // 1/sqrt(128)
  int qtop = qr0 + ((lane >> 4) << 3);        // base row for this lane half

  int ktiles = (qb + 64) >> 5;                // causal: k0 < qb+64

  // prologue: start DMA of K tile 0
  if (leaderWave) tdm_load_k_tile(K + bh, ldsK[0]);

  for (int kt = 0; kt < ktiles; ++kt) {
    const int cur = kt & 1;
    int k0 = kt * 32;
    __syncthreads();     // prior readers of ldsK[cur^1] / ldsVt are done

    // V tile transposed into [d][kk] (manual: TDM cannot transpose)
#pragma unroll
    for (int i = 0; i < 4; ++i) {
      int idx = tid + i * 128;
      int row = idx >> 4, c = (idx & 15) << 3;
      uint4 raw = *(const uint4*)(V + bh + (size_t)(k0 + row) * DMODEL + c);
      const bf16* e = (const bf16*)&raw;
#pragma unroll
      for (int j = 0; j < 8; ++j) ldsVt[(c + j) * 32 + row] = e[j];
    }

    // kick DMA for tile kt+1 into the other buffer; then ensure tile kt landed
    if (leaderWave) {
      if (kt + 1 < ktiles) {
        tdm_load_k_tile(K + bh + (size_t)(kt + 1) * 32 * DMODEL, ldsK[cur ^ 1]);
        __builtin_amdgcn_s_wait_tensorcnt(1);   // in-order: tile kt complete
      } else {
        __builtin_amdgcn_s_wait_tensorcnt(0);
      }
    }
    __syncthreads();                            // publish K tile + V tile

    // S = q * K^T : batch 4 fragment loads per 16-col group to keep >=4
    // independent ds_load_b128 pairs in flight ahead of each WMMA burst.
    v8f sacc[2] = {};
#pragma unroll
    for (int nt = 0; nt < 2; ++nt) {
      v16bf kb[4];
#pragma unroll
      for (int dk = 0; dk < 4; ++dk)
        kb[dk] = load_frag(ldsK[cur], 128, nt * 16, dk * 32);
#pragma unroll
      for (int dk = 0; dk < 4; ++dk)
        sacc[nt] = wmma_bf16(qa[dk], kb[dk], sacc[nt]);
    }

    // online softmax + write P (bf16) to per-wave LDS scratch
    int kcol = k0 + (lane & 15);
#pragma unroll
    for (int r = 0; r < 8; ++r) {
      int qrow = qtop + r;
      float s0 = sacc[0][r] * scale;
      float s1 = sacc[1][r] * scale;
      if (kcol > qrow)      s0 = -1.0e30f;    // causal mask
      if (kcol + 16 > qrow) s1 = -1.0e30f;
      float mx   = half_red_max(fmaxf(s0, s1));
      float mnew = fmaxf(mrow[r], mx);
      float alpha = __expf(mrow[r] - mnew);
      float p0 = __expf(s0 - mnew);
      float p1 = __expf(s1 - mnew);
      lrow[r] = lrow[r] * alpha + half_red_sum(p0 + p1);
      mrow[r] = mnew;
      int prow = ((lane >> 4) << 3) + r;
      int pcol = lane & 15;
      ldsP[wid * 512 + prow * 32 + pcol]      = (bf16)p0;
      ldsP[wid * 512 + prow * 32 + 16 + pcol] = (bf16)p1;
#pragma unroll
      for (int dt = 0; dt < 8; ++dt) oacc[dt][r] *= alpha;
    }

    // O += P(16x32) * V(32x128): one P A-frag, 8 V B-frags (batched 4-deep)
    v16bf pa = load_frag(ldsP + wid * 512, 32, 0, 0);
#pragma unroll
    for (int g = 0; g < 2; ++g) {
      v16bf vb[4];
#pragma unroll
      for (int j = 0; j < 4; ++j)
        vb[j] = load_frag(ldsVt, 32, (g * 4 + j) * 16, 0);
#pragma unroll
      for (int j = 0; j < 4; ++j)
        oacc[g * 4 + j] = wmma_bf16(pa, vb[j], oacc[g * 4 + j]);
    }
  }

  // epilogue: normalize and store bf16 (feeds the final x@Wo^T GEMM)
#pragma unroll
  for (int r = 0; r < 8; ++r) {
    float inv = 1.0f / lrow[r];
    int qrow = qtop + r;
#pragma unroll
    for (int dt = 0; dt < 8; ++dt) {
      int d = dt * 16 + (lane & 15);
      O[bh + (size_t)qrow * DMODEL + d] = (bf16)(oacc[dt][r] * inv);
    }
  }
}

// ---------------------------------------------------------------------------
// launcher
// ---------------------------------------------------------------------------
extern "C" void kernel_launch(void* const* d_in, const int* in_sizes, int n_in,
                              void* d_out, int out_size, void* d_ws, size_t ws_size,
                              hipStream_t stream) {
  const float* x    = (const float*)d_in[0];
  // d_in[1] = start_pos (0), d_in[4] = mask (causal handled in-kernel)
  const float* fcos = (const float*)d_in[2];
  const float* fsin = (const float*)d_in[3];
  const float* wq   = (const float*)d_in[5];
  const float* wk   = (const float*)d_in[6];
  const float* wv   = (const float*)d_in[7];
  const float* wo   = (const float*)d_in[8];

  bf16* ws  = (bf16*)d_ws;
  bf16* xbf = ws + (size_t)NELEM * 0;
  bf16* wqb = ws + (size_t)NELEM * 1;
  bf16* wkb = ws + (size_t)NELEM * 2;
  bf16* wvb = ws + (size_t)NELEM * 3;
  bf16* wob = ws + (size_t)NELEM * 4;
  bf16* xq  = ws + (size_t)NELEM * 5;
  bf16* xk  = ws + (size_t)NELEM * 6;
  bf16* xv  = ws + (size_t)NELEM * 7;
  bf16* ao  = ws + (size_t)NELEM * 8;

  const int n8 = NELEM / 8;
  const int cvtBlocks = n8 / 256;
  cvt_f32_bf16<<<cvtBlocks, 256, 0, stream>>>(x,  xbf, n8);
  cvt_f32_bf16<<<cvtBlocks, 256, 0, stream>>>(wq, wqb, n8);
  cvt_f32_bf16<<<cvtBlocks, 256, 0, stream>>>(wk, wkb, n8);
  cvt_f32_bf16<<<cvtBlocks, 256, 0, stream>>>(wv, wvb, n8);
  cvt_f32_bf16<<<cvtBlocks, 256, 0, stream>>>(wo, wob, n8);

  dim3 gg(DMODEL / 128, (B_SZ * S_LEN) / 128);    // (32, 32)
  gemm_xwT<bf16><<<gg, 256, 0, stream>>>(xbf, wqb, xq, B_SZ * S_LEN, DMODEL, DMODEL);
  gemm_xwT<bf16><<<gg, 256, 0, stream>>>(xbf, wkb, xk, B_SZ * S_LEN, DMODEL, DMODEL);
  gemm_xwT<bf16><<<gg, 256, 0, stream>>>(xbf, wvb, xv, B_SZ * S_LEN, DMODEL, DMODEL);

  const int ropeBlocks = (B_SZ * S_LEN * NHEAD * (HDIM / 2)) / 256;  // 32768
  rope_bf16<<<ropeBlocks, 256, 0, stream>>>(xq, fcos, fsin);
  rope_bf16<<<ropeBlocks, 256, 0, stream>>>(xk, fcos, fsin);

  flash_attn<<<B_SZ * NHEAD * (S_LEN / 64), 128, 0, stream>>>(xq, xk, xv, ao);

  gemm_xwT<float><<<gg, 256, 0, stream>>>(ao, wob, (float*)d_out,
                                          B_SZ * S_LEN, DMODEL, DMODEL);
}
